// CFConv_8821862826243
// MI455X (gfx1250) — matrix operand order repaired and checked by reading, not compiled, not run on hardware
//
#include <hip/hip_runtime.h>
#include <hip/hip_bf16.h>

// ---------------------------------------------------------------------------
// CFConv for MI455X (gfx1250, wave32, WMMA).
//   rbf  : [16 rows x 320 K] f16 tile per (b, i, j-tile)   (K padded 300->320)
//   GEMM1: 16x320 @ 320x64  -> 16x64  (10x v_wmma_f32_16x16x32_f16 per wave,
//                                      dual accumulator chains)
//   GEMM2: 16x64  @ 64x64   -> 16x64  ( 2x v_wmma per wave)
//   out[b,j,c] += h[b,i,c] * f[b,i,j,c]  reduced over i via f32 atomics.
// ---------------------------------------------------------------------------

typedef __attribute__((ext_vector_type(16))) _Float16 v16h;
typedef __attribute__((ext_vector_type(8)))  _Float16 v8h;
typedef __attribute__((ext_vector_type(4)))  _Float16 v4h;
typedef __attribute__((ext_vector_type(8)))  float    v8f;

#define NUM_RBF   300
#define KPAD      320          // 10 k-slices of 32
#define RSTRIDE   328          // LDS row stride (halves): 656B, 8B/16B-aligned
#define HD        64
#define W2STRIDE  72           // 144B rows, 16B-aligned
#define FSTRIDE   72

__device__ __forceinline__ float softplus_f(float x) {
    // stable: max(x,0) + log(1 + exp(-|x|)); lowers to v_exp_f32 / v_log_f32
    return fmaxf(x, 0.0f) + __logf(1.0f + __expf(-fabsf(x)));
}

__device__ __forceinline__ v16h cat8(v8h lo, v8h hi) {
    return __builtin_shufflevector(lo, hi, 0, 1, 2, 3, 4, 5, 6, 7,
                                           8, 9, 10, 11, 12, 13, 14, 15);
}

__global__ void __launch_bounds__(128)
cfconv_zero_kernel(float* __restrict__ out, int n) {
    int i = blockIdx.x * blockDim.x + threadIdx.x;
    if (i < n) out[i] = 0.0f;
}

__global__ void __launch_bounds__(128)
cfconv_wmma_kernel(const float* __restrict__ hmat,
                   const float* __restrict__ dist,
                   const float* __restrict__ W1,
                   const float* __restrict__ b1,
                   const float* __restrict__ W2,
                   const float* __restrict__ b2,
                   float* __restrict__ out) {
    // LDS: transposed fp16 weights [c][k], RBF A-tile, F exchange buffer
    __shared__ __align__(16) _Float16 w1t[HD * RSTRIDE];   // ~41 KB
    __shared__ __align__(16) _Float16 w2t[HD * W2STRIDE];  // ~9 KB
    __shared__ __align__(16) _Float16 arbf[16 * RSTRIDE];  // ~10.5 KB
    __shared__ __align__(16) _Float16 fbuf[16 * FSTRIDE];  // ~2.3 KB
    __shared__ __align__(16) float    distsh[16 * 16];
    __shared__ __align__(16) float    hsh[16 * HD];

    const int tid  = threadIdx.x;
    const int lane = tid & 31;
    const int wave = tid >> 5;        // 0..3 : N-block owner
    const int hi   = lane >> 4;       // half-wave select
    const int lm   = lane & 15;       // A/C row  (or B/C column)
    const int n0   = wave * 16;

    const int blk = blockIdx.x;       // 512 blocks
    const int b   = blk >> 4;                  // batch
    const int j0  = ((blk >> 2) & 3) * 16;     // output-atom tile
    const int i0  = (blk & 3) * 16;            // input-atom chunk

    // ---- zero padded LDS weight buffers with 16B vector stores ----
    {
        const v8h z = {};
        for (int q = tid; q < (HD * RSTRIDE) / 8; q += 128)
            *(v8h*)&w1t[q * 8] = z;
        for (int q = tid; q < (HD * W2STRIDE) / 8; q += 128)
            *(v8h*)&w2t[q * 8] = z;
    }
    __syncthreads();

    // ---- transpose + convert weights (coalesced global, vector LDS) ----
    // chunk q: channel c = q & 63, rows r0..r0+3. Threads walk c fastest ->
    // each of the 4 global loads is fully coalesced; LDS store is one b64.
    for (int q = tid; q < (NUM_RBF / 4) * HD; q += 128) {
        int c = q & 63, r0 = (q >> 6) * 4;
        v4h o;
#pragma unroll
        for (int e = 0; e < 4; ++e) o[e] = (_Float16)W1[(r0 + e) * HD + c];
        *(v4h*)&w1t[c * RSTRIDE + r0] = o;
    }
    for (int q = tid; q < (HD / 4) * HD; q += 128) {
        int c = q & 63, r0 = (q >> 6) * 4;
        v4h o;
#pragma unroll
        for (int e = 0; e < 4; ++e) o[e] = (_Float16)W2[(r0 + e) * HD + c];
        *(v4h*)&w2t[c * W2STRIDE + r0] = o;
    }
    for (int idx = tid; idx < 16 * 16; idx += 128) {
        int ii = idx >> 4, m = idx & 15;
        distsh[idx] = dist[((b * 64) + i0 + ii) * 64 + j0 + m];
    }
    for (int idx = tid; idx < 16 * HD; idx += 128) {
        int ii = idx >> 6, c = idx & 63;
        hsh[idx] = hmat[((b * 64) + i0 + ii) * 64 + c];
    }
    __syncthreads();

    // ---- loop-invariant B fragments in registers ----
    // B layout (32x16 f16): lanes 0-15 K={0..7,16..23}, lanes 16-31 K={8..15,24..31}.
    const int kselB = hi ? 8 : 0;
    v16h b1f[10];
#pragma unroll
    for (int ks = 0; ks < 10; ++ks) {
        const _Float16* p = &w1t[(n0 + lm) * RSTRIDE + ks * 32 + kselB];
        b1f[ks] = cat8(*(const v8h*)p, *(const v8h*)(p + 16));
    }
    v16h b2f[2];
#pragma unroll
    for (int ks = 0; ks < 2; ++ks) {
        const _Float16* p = &w2t[(n0 + lm) * W2STRIDE + ks * 32 + kselB];
        b2f[ks] = cat8(*(const v8h*)p, *(const v8h*)(p + 16));
    }
    const float b1v = b1[n0 + lm];
    const float b2v = b2[n0 + lm];
    // A layout (16x32 f16): lanes 0-15 K={0..7,16..23}, lanes 16-31 K={8..15,24..31}
    const int kselA = hi ? 8 : 0;

    // RBF-fill ownership: row m = tid>>3, k chunks (tid&7)*8 + 64*it, it<5.
    const int fill_m  = tid >> 3;
    const int fill_kc = (tid & 7) * 8;

    v8f outacc = {};

    for (int ii = 0; ii < 16; ++ii) {
        // -- phase 1: cooperative RBF tile, 5x v8h stores per thread --
        // note: k>=300 gives exp(-10*(d-30+)^2) <= exp(-4000) == +0.0f for
        // d in [0,10], so the pad region self-zeroes; no predicate needed.
        {
            const float d = distsh[ii * 16 + fill_m];
#pragma unroll
            for (int it = 0; it < 5; ++it) {
                const int k0 = fill_kc + it * 64;
                v8h o;
#pragma unroll
                for (int e = 0; e < 8; ++e) {
                    float t = d - 0.1f * (float)(k0 + e);
                    o[e] = (_Float16)__expf(-10.0f * t * t);
                }
                *(v8h*)&arbf[fill_m * RSTRIDE + k0] = o;
            }
        }
        __syncthreads();   // arbf ready; also fences prev-iter fbuf reads

        // -- phase 2: GEMM1 (K=320), two independent accumulator chains --
        v8f acc_a = {}, acc_b = {};
#pragma unroll
        for (int ks = 0; ks < 10; ks += 2) {
            const _Float16* pa = &arbf[lm * RSTRIDE + ks * 32 + kselA];
            v16h a0 = cat8(*(const v8h*)pa, *(const v8h*)(pa + 16));
            const _Float16* pb = pa + 32;
            v16h a1 = cat8(*(const v8h*)pb, *(const v8h*)(pb + 16));
            acc_a = __builtin_amdgcn_wmma_f32_16x16x32_f16(
                false, a0, false, b1f[ks], (short)0, acc_a, false, false);
            acc_b = __builtin_amdgcn_wmma_f32_16x16x32_f16(
                false, a1, false, b1f[ks + 1], (short)0, acc_b, false, false);
        }
#pragma unroll
        for (int v = 0; v < 8; ++v) {
            int j = v + 8 * hi;      // C layout: VGPR v, half-wave hi -> row
            float x = acc_a[v] + acc_b[v] + b1v;
            fbuf[j * FSTRIDE + n0 + lm] = (_Float16)softplus_f(x);
        }
        __syncthreads();   // fbuf ready; also fences this-iter arbf reads

        // -- phase 3: GEMM2 (K=64) + softplus, scale by h, accumulate --
        v8f acc2 = {};
#pragma unroll
        for (int ks = 0; ks < 2; ++ks) {
            const _Float16* p = &fbuf[lm * FSTRIDE + ks * 32 + kselA];
            v16h a = cat8(*(const v8h*)p, *(const v8h*)(p + 16));
            acc2 = __builtin_amdgcn_wmma_f32_16x16x32_f16(
                false, a, false, b2f[ks], (short)0, acc2, false, false);
        }
        const float hv = hsh[ii * HD + n0 + lm];
#pragma unroll
        for (int v = 0; v < 8; ++v) {
            outacc[v] += hv * softplus_f(acc2[v] + b2v);
        }
        // no barrier here: next-iter arbf writes are fenced by the barrier
        // after phase 2; next-iter fbuf writes by the barrier after phase 1.
    }

    // ---- reduce partial i-chunks into out via f32 atomics ----
    const int cg = n0 + lm;
#pragma unroll
    for (int v = 0; v < 8; ++v) {
        int j = j0 + v + 8 * hi;
        atomicAdd(&out[(b * 64 + j) * 64 + cg], outacc[v]);
    }
}

extern "C" void kernel_launch(void* const* d_in, const int* in_sizes, int n_in,
                              void* d_out, int out_size, void* d_ws, size_t ws_size,
                              hipStream_t stream) {
    const float* h    = (const float*)d_in[0];
    const float* dist = (const float*)d_in[1];
    const float* W1   = (const float*)d_in[2];
    const float* b1   = (const float*)d_in[3];
    const float* W2   = (const float*)d_in[4];
    const float* b2   = (const float*)d_in[5];
    float* out = (float*)d_out;

    cfconv_zero_kernel<<<(out_size + 127) / 128, 128, 0, stream>>>(out, out_size);
    // 32 batches x 4 j-tiles x 4 i-chunks = 512 workgroups, 4 waves each
    cfconv_wmma_kernel<<<512, 128, 0, stream>>>(h, dist, W1, b1, W2, b2, out);
}
